// SSD_Block_53369263620388
// MI455X (gfx1250) — matrix operand order repaired and checked by reading, not compiled, not run on hardware
//
#include <hip/hip_runtime.h>
#include <hip/hip_bf16.h>

// ---------------------------------------------------------------------------
// Mamba2 SSD block for MI455X (gfx1250), wave32 + WMMA bf16x3 GEMMs.
// GEMMs use V_WMMA_F32_16X16X32_BF16 with error-compensated fp32 emulation
// (A = A_hi + A_lo in bf16; D = Ahi*Bhi + Ahi*Blo + Alo*Bhi).
// Each wave computes a 16x64 output strip (4 N-tiles); GEMM is templated on
// K so all fragment addresses are immediate offsets from loop-invariant
// per-lane base pointers (no per-iteration 64-bit address VALU -> no
// WMMA->VALU hazard NOP sites in the hot loop).
// ---------------------------------------------------------------------------

typedef __attribute__((ext_vector_type(16))) __bf16        v16bf;
typedef __attribute__((ext_vector_type(8)))  float         v8f;
typedef __attribute__((ext_vector_type(8)))  unsigned int  v8u;
typedef __attribute__((ext_vector_type(4)))  unsigned int  v4u;

#define WG 256
#define NT 4            // N-tiles per wave (16 cols each -> 64-wide strip)

__device__ __forceinline__ unsigned short f2bf(float f) {
  unsigned u = __float_as_uint(f);
  unsigned r = u + 0x7FFFu + ((u >> 16) & 1u);   // round-to-nearest-even
  return (unsigned short)(r >> 16);
}
__device__ __forceinline__ float bf2f(unsigned short h) {
  return __uint_as_float(((unsigned)h) << 16);
}
__device__ __forceinline__ float sigm(float x)  { return 1.0f / (1.0f + __expf(-x)); }
__device__ __forceinline__ float splus(float x) { return (x > 20.0f) ? x : log1pf(__expf(x)); }

// ---------------------------------------------------------------------------
// LayerNorm over rows of length Dm, output split into bf16 hi/lo planes
// ---------------------------------------------------------------------------
__global__ void ln_split_kernel(const float* __restrict__ X,
                                const float* __restrict__ w,
                                const float* __restrict__ bia,
                                unsigned short* __restrict__ Ahi,
                                unsigned short* __restrict__ Alo,
                                int R, int Dm) {
  __shared__ float red[WG];
  int row = blockIdx.x;
  if (row >= R) return;
  int tid = threadIdx.x;
  const float* xr = X + (size_t)row * Dm;
  float s = 0.f, s2 = 0.f;
  for (int c = tid; c < Dm; c += WG) { float v = xr[c]; s += v; s2 += v * v; }
  red[tid] = s; __syncthreads();
  for (int o = WG / 2; o > 0; o >>= 1) { if (tid < o) red[tid] += red[tid + o]; __syncthreads(); }
  float mean = red[0] / (float)Dm;
  __syncthreads();
  red[tid] = s2; __syncthreads();
  for (int o = WG / 2; o > 0; o >>= 1) { if (tid < o) red[tid] += red[tid + o]; __syncthreads(); }
  float var = red[0] / (float)Dm - mean * mean;
  float inv = rsqrtf(var + 1e-5f);
  for (int c = tid; c < Dm; c += WG) {
    float v = (xr[c] - mean) * inv * w[c] + bia[c];
    unsigned short hi = f2bf(v);
    unsigned short lo = f2bf(v - bf2f(hi));
    size_t o = (size_t)row * Dm + c;
    Ahi[o] = hi; Alo[o] = lo;
  }
}

// ---------------------------------------------------------------------------
// Split fp32 weight W (K x N, row major) into transposed bf16 hi/lo
// (Npad x K), zero-filling pad rows n in [N, Npad).
// ---------------------------------------------------------------------------
__global__ void split_wT_kernel(const float* __restrict__ W,
                                unsigned short* __restrict__ Thi,
                                unsigned short* __restrict__ Tlo,
                                int K, int N, int Npad) {
  size_t idx = (size_t)blockIdx.x * blockDim.x + threadIdx.x;
  size_t tot = (size_t)Npad * K;
  if (idx >= tot) return;
  int n = (int)(idx / K), k = (int)(idx % K);
  float v = (n < N) ? W[(size_t)k * N + n] : 0.0f;
  unsigned short hi = f2bf(v);
  unsigned short lo = f2bf(v - bf2f(hi));
  Thi[idx] = hi; Tlo[idx] = lo;
}

// ---------------------------------------------------------------------------
// WMMA fragments (CDNA5 16-bit layouts). Lane-dependent base folded into the
// pointer; each fragment is two aligned 16B loads at constant offsets.
// A 16x32 (per lane row m=lane&15, +8 elem if lane>=16): bytes +0 and +32
// B 32x16 (per lane col n=lane&15, +8 elem if lane>=16): bytes +0 and +32
// ---------------------------------------------------------------------------
__device__ __forceinline__ v16bf frag_from2(v4u lo, v4u hi) {
  v8u v;
  v[0] = lo[0]; v[1] = lo[1]; v[2] = lo[2]; v[3] = lo[3];
  v[4] = hi[0]; v[5] = hi[1]; v[6] = hi[2]; v[7] = hi[3];
  return __builtin_bit_cast(v16bf, v);
}
__device__ __forceinline__ v16bf frag_a16(const unsigned short* __restrict__ p) {
  return frag_from2(*(const v4u*)p, *(const v4u*)(p + 16));
}
__device__ __forceinline__ v16bf frag_b16(const unsigned short* __restrict__ p) {
  return frag_from2(*(const v4u*)p, *(const v4u*)(p + 16));
}

// ---------------------------------------------------------------------------
// C = A(MxK) @ B(KxN) in ~fp32 via bf16x3; BT is B transposed + padded
// (Npad x K, Npad multiple of 16*NT, pad rows zero). One 16x64 strip per
// wave, 8 waves (128 M-rows) per workgroup. Optional fused residual add.
// ---------------------------------------------------------------------------
template <int K>
__global__ void gemm_bf16x3_kernel(const unsigned short* __restrict__ Ahi,
                                   const unsigned short* __restrict__ Alo,
                                   const unsigned short* __restrict__ BThi,
                                   const unsigned short* __restrict__ BTlo,
                                   float* __restrict__ C,
                                   const float* __restrict__ Res,
                                   int M, int N) {
  int lane = threadIdx.x & 31;
  int wave = threadIdx.x >> 5;
  int n0 = blockIdx.x * (16 * NT);
  int m0 = blockIdx.y * 128 + wave * 16;
  if (m0 >= M) return;

  int lhalf = lane >> 4;
  const unsigned short* pah = Ahi + (size_t)(m0 + (lane & 15)) * K + lhalf * 8;
  const unsigned short* pal = Alo + (size_t)(m0 + (lane & 15)) * K + lhalf * 8;
  const unsigned short* pbh = BThi + (size_t)(n0 + (lane & 15)) * K + lhalf * 8;
  const unsigned short* pbl = BTlo + (size_t)(n0 + (lane & 15)) * K + lhalf * 8;

  v8f acc[NT];
#pragma unroll
  for (int nt = 0; nt < NT; ++nt) acc[nt] = v8f{};

#pragma unroll 2
  for (int k0 = 0; k0 < K; k0 += 32) {
    v16bf ah = frag_a16(pah + k0);
    v16bf al = frag_a16(pal + k0);
#pragma unroll
    for (int nt = 0; nt < NT; ++nt) {
      v16bf bh = frag_b16(pbh + k0 + nt * (16 * K));   // constant tile stride
      v16bf bl = frag_b16(pbl + k0 + nt * (16 * K));
      acc[nt] = __builtin_amdgcn_wmma_f32_16x16x32_bf16(false, ah, false, bh, (short)0, acc[nt], false, false);
      acc[nt] = __builtin_amdgcn_wmma_f32_16x16x32_bf16(false, ah, false, bl, (short)0, acc[nt], false, false);
      acc[nt] = __builtin_amdgcn_wmma_f32_16x16x32_bf16(false, al, false, bh, (short)0, acc[nt], false, false);
    }
    if (k0 + 32 < K) {
      __builtin_prefetch(pah + k0 + 32, 0, 1);   // global_prefetch_b8
      __builtin_prefetch(pbh + k0 + 32, 0, 1);
    }
  }

  int rbase = m0 + lhalf * 8;
  if (Res) {
#pragma unroll
    for (int nt = 0; nt < NT; ++nt) {
      int col = n0 + nt * 16 + (lane & 15);
      if (col < N) {
#pragma unroll
        for (int r = 0; r < 8; ++r) {
          size_t o = (size_t)(rbase + r) * N + col;
          C[o] = acc[nt][r] + Res[o];
        }
      }
    }
  } else {
#pragma unroll
    for (int nt = 0; nt < NT; ++nt) {
      int col = n0 + nt * 16 + (lane & 15);
      if (col < N) {
#pragma unroll
        for (int r = 0; r < 8; ++r) {
          size_t o = (size_t)(rbase + r) * N + col;
          C[o] = acc[nt][r];
        }
      }
    }
  }
}

// ---------------------------------------------------------------------------
// Causal depthwise conv (width 4) + SiLU on the xBC slice of zxbcdt
// ---------------------------------------------------------------------------
__global__ void conv_silu_kernel(const float* __restrict__ Z,
                                 const float* __restrict__ cw,
                                 const float* __restrict__ cb,
                                 float* __restrict__ XBC,
                                 int B, int L, int din, int dinner, int convdim) {
  size_t idx = (size_t)blockIdx.x * blockDim.x + threadIdx.x;
  size_t tot = (size_t)B * L * convdim;
  if (idx >= tot) return;
  int c = (int)(idx % convdim);
  size_t bl = idx / convdim;
  int t = (int)(bl % L);
  int b = (int)(bl / L);
  float acc = cb[c];
#pragma unroll
  for (int k = 0; k < 4; ++k) {
    int ts = t + k - 3;
    if (ts >= 0)
      acc += cw[k * convdim + c] * Z[((size_t)(b * L + ts)) * din + dinner + c];
  }
  XBC[idx] = acc * sigm(acc);
}

// ---------------------------------------------------------------------------
// dt = softplus(dt_raw + bias); da = exp(dt * -exp(A_log))
// ---------------------------------------------------------------------------
__global__ void dtda_kernel(const float* __restrict__ Z,
                            const float* __restrict__ dtb,
                            const float* __restrict__ Alog,
                            float* __restrict__ DT, float* __restrict__ DA,
                            int BL, int din, int off, int H) {
  int idx = blockIdx.x * blockDim.x + threadIdx.x;
  if (idx >= BL * H) return;
  int h = idx % H;
  int r = idx / H;
  float dt = splus(Z[(size_t)r * din + off + h] + dtb[h]);
  DT[idx] = dt;
  DA[idx] = __expf(dt * (-__expf(Alog[h])));
}

// ---------------------------------------------------------------------------
// Sequential SSM scan: one workgroup per (batch, head); state h[64][64] in
// registers (16 f32/thread), B_t/C_t staged in LDS each step.
// ---------------------------------------------------------------------------
__global__ void scan_kernel(const float* __restrict__ XBC,
                            const float* __restrict__ DT,
                            const float* __restrict__ DA,
                            const float* __restrict__ Dh,
                            float* __restrict__ Y,
                            int B, int L, int H, int convdim, int dinner) {
  __shared__ float sB[64], sC[64], yred[WG];
  int b = blockIdx.x / H, hh = blockIdx.x % H;
  int tid = threadIdx.x;
  int p = tid >> 2, nq = tid & 3, n0 = nq * 16;
  float hst[16];
#pragma unroll
  for (int i = 0; i < 16; ++i) hst[i] = 0.f;

  for (int t = 0; t < L; ++t) {
    size_t row = (size_t)(b * L + t);
    const float* xb = XBC + row * convdim;
    if (tid < 128) {
      if (tid < 64) sB[tid] = xb[dinner + tid];
      else          sC[tid - 64] = xb[dinner + tid];
    }
    __syncthreads();
    float dtv = DT[row * H + hh];
    float dav = DA[row * H + hh];
    float xv  = xb[hh * 64 + p];
    float dtx = dtv * xv;
    float yp = 0.f;
#pragma unroll
    for (int i = 0; i < 16; ++i) {
      hst[i] = dav * hst[i] + dtx * sB[n0 + i];
      yp += hst[i] * sC[n0 + i];
    }
    yred[tid] = yp;
    __syncthreads();
    if (nq == 0) {
      float y = yred[tid] + yred[tid + 1] + yred[tid + 2] + yred[tid + 3] + Dh[hh] * xv;
      Y[row * dinner + hh * 64 + p] = y;
    }
  }
}

// ---------------------------------------------------------------------------
// y *= silu(z); RMSNorm(y)*rms_w; split into bf16 hi/lo for W_out GEMM
// ---------------------------------------------------------------------------
__global__ void gate_rms_split_kernel(const float* __restrict__ Z,
                                      float* __restrict__ Y,
                                      const float* __restrict__ rw,
                                      unsigned short* __restrict__ Yhi,
                                      unsigned short* __restrict__ Ylo,
                                      int din, int dinner) {
  __shared__ float red[WG];
  int row = blockIdx.x, tid = threadIdx.x;
  const float* zr = Z + (size_t)row * din;
  float* yr = Y + (size_t)row * dinner;
  float s2 = 0.f;
  for (int c = tid; c < dinner; c += WG) {
    float z = zr[c];
    float y = yr[c] * z * sigm(z);
    yr[c] = y;
    s2 += y * y;
  }
  red[tid] = s2; __syncthreads();
  for (int o = WG / 2; o > 0; o >>= 1) { if (tid < o) red[tid] += red[tid + o]; __syncthreads(); }
  float inv = rsqrtf(red[0] / (float)dinner + 1e-5f);
  for (int c = tid; c < dinner; c += WG) {
    float v = yr[c] * inv * rw[c];
    unsigned short hi = f2bf(v);
    unsigned short lo = f2bf(v - bf2f(hi));
    size_t o = (size_t)row * dinner + c;
    Yhi[o] = hi; Ylo[o] = lo;
  }
}

// ---------------------------------------------------------------------------
// Transposes: (B,L,C) -> (B,C,L), and transpose-add epilogue
// ---------------------------------------------------------------------------
__global__ void transpose_kernel(const float* __restrict__ X, float* __restrict__ T,
                                 int B, int L, int C) {
  size_t idx = (size_t)blockIdx.x * blockDim.x + threadIdx.x;
  size_t tot = (size_t)B * L * C;
  if (idx >= tot) return;
  int l = (int)(idx % L);
  int c = (int)((idx / L) % C);
  int b = (int)(idx / ((size_t)L * C));
  T[idx] = X[((size_t)b * L + l) * C + c];
}

__global__ void transpose_add_kernel(const float* __restrict__ G,   // (B,C,L)
                                     const float* __restrict__ Xs,  // (B,L,C)
                                     float* __restrict__ Out,       // (B,L,C)
                                     int B, int L, int C) {
  size_t idx = (size_t)blockIdx.x * blockDim.x + threadIdx.x;
  size_t tot = (size_t)B * L * C;
  if (idx >= tot) return;
  int c = (int)(idx % C);
  int l = (int)((idx / C) % L);
  int b = (int)(idx / ((size_t)L * C));
  Out[idx] = G[((size_t)b * C + c) * L + l] + Xs[idx];
}

// ---------------------------------------------------------------------------
// Host orchestration
// ---------------------------------------------------------------------------
static inline size_t alignup256(size_t x) { return (x + 255) & ~(size_t)255; }
static inline int cdiv(long long a, long long b) { return (int)((a + b - 1) / b); }

extern "C" void kernel_launch(void* const* d_in, const int* in_sizes, int n_in,
                              void* d_out, int out_size, void* d_ws, size_t ws_size,
                              hipStream_t stream) {
  (void)in_sizes; (void)n_in; (void)out_size; (void)ws_size;

  const float* x         = (const float*)d_in[0];
  const float* ln_spa_w  = (const float*)d_in[1];
  const float* ln_spa_b  = (const float*)d_in[2];
  const float* ln_spe_w  = (const float*)d_in[3];
  const float* ln_spe_b  = (const float*)d_in[4];

  struct MP {
    int B, Lseq, dmodel, dinner, H, convdim, din;
    const float *W_in, *conv_w, *conv_b, *dt_bias, *A_log, *Dh, *rms_w, *W_out;
    const float *ln_w, *ln_b;
  };

  MP spa = { 8, 1024, 128, 256, 4, 384, 644,
             (const float*)d_in[5],  (const float*)d_in[6],  (const float*)d_in[7],
             (const float*)d_in[8],  (const float*)d_in[9],  (const float*)d_in[10],
             (const float*)d_in[11], (const float*)d_in[12],
             ln_spa_w, ln_spa_b };
  MP spe = { 8, 128, 1024, 2048, 32, 2176, 4256,
             (const float*)d_in[13], (const float*)d_in[14], (const float*)d_in[15],
             (const float*)d_in[16], (const float*)d_in[17], (const float*)d_in[18],
             (const float*)d_in[19], (const float*)d_in[20],
             ln_spe_w, ln_spe_b };

  char* ws = (char*)d_ws;
  size_t off = 0;
  auto alloc = [&](size_t bytes) -> void* {
    void* p = ws + off;
    off = alignup256(off + bytes);
    return p;
  };

  const int BTOT = 8, LTOT = 1024, CTOT = 128;
  const size_t NX = (size_t)BTOT * LTOT * CTOT;
  const int STRIP = 16 * NT;   // 64

  // K is one of {128, 256, 1024, 2048}; dispatch templated GEMM
  auto launch_gemm = [&](int K, const unsigned short* ahi, const unsigned short* alo,
                         const unsigned short* bhi, const unsigned short* blo,
                         float* Cp, const float* Resp, int M, int N, int nstrips) {
    dim3 g(nstrips, M / 128), b(WG);
    switch (K) {
      case 128:  gemm_bf16x3_kernel<128><<<g, b, 0, stream>>>(ahi, alo, bhi, blo, Cp, Resp, M, N); break;
      case 256:  gemm_bf16x3_kernel<256><<<g, b, 0, stream>>>(ahi, alo, bhi, blo, Cp, Resp, M, N); break;
      case 1024: gemm_bf16x3_kernel<1024><<<g, b, 0, stream>>>(ahi, alo, bhi, blo, Cp, Resp, M, N); break;
      case 2048: gemm_bf16x3_kernel<2048><<<g, b, 0, stream>>>(ahi, alo, bhi, blo, Cp, Resp, M, N); break;
      default: break; // unused
    }
  };

  float* x_spa_final = (float*)alloc(NX * 4);   // persistent across phases
  size_t phase_base = off;

  auto run_mamba = [&](const MP& m, const float* src, float* dst, const float* resid) {
    int rows = m.B * m.Lseq;                      // tokens (multiple of 128)
    int NpadIn  = cdiv(m.din, STRIP) * STRIP;     // padded N for W_in^T
    int NpadOut = cdiv(m.dmodel, STRIP) * STRIP;  // padded N for W_out^T

    unsigned short* a_hi  = (unsigned short*)alloc((size_t)rows * m.dmodel * 2);
    unsigned short* a_lo  = (unsigned short*)alloc((size_t)rows * m.dmodel * 2);
    unsigned short* wi_hi = (unsigned short*)alloc((size_t)NpadIn * m.dmodel * 2);
    unsigned short* wi_lo = (unsigned short*)alloc((size_t)NpadIn * m.dmodel * 2);
    float*          zx    = (float*)alloc((size_t)rows * m.din * 4);
    float*          xbc   = (float*)alloc((size_t)rows * m.convdim * 4);
    float*          dt    = (float*)alloc((size_t)rows * m.H * 4);
    float*          da    = (float*)alloc((size_t)rows * m.H * 4);
    float*          ybuf  = (float*)alloc((size_t)rows * m.dinner * 4);
    unsigned short* yn_hi = (unsigned short*)alloc((size_t)rows * m.dinner * 2);
    unsigned short* yn_lo = (unsigned short*)alloc((size_t)rows * m.dinner * 2);
    unsigned short* wo_hi = (unsigned short*)alloc((size_t)NpadOut * m.dinner * 2);
    unsigned short* wo_lo = (unsigned short*)alloc((size_t)NpadOut * m.dinner * 2);

    // 1) LayerNorm + bf16 hi/lo split
    ln_split_kernel<<<dim3(rows), dim3(WG), 0, stream>>>(
        src, m.ln_w, m.ln_b, a_hi, a_lo, rows, m.dmodel);
    // 2) split W_in transposed (padded to NpadIn rows)
    split_wT_kernel<<<dim3(cdiv((long long)NpadIn * m.dmodel, WG)), dim3(WG), 0, stream>>>(
        m.W_in, wi_hi, wi_lo, m.dmodel, m.din, NpadIn);
    // 3) zxbcdt = x_ln @ W_in   (M=rows, N=din, K=dmodel)
    launch_gemm(m.dmodel, a_hi, a_lo, wi_hi, wi_lo, zx, nullptr, rows, m.din, NpadIn / STRIP);
    // 4) conv + SiLU
    conv_silu_kernel<<<dim3(cdiv((long long)rows * m.convdim, WG)), dim3(WG), 0, stream>>>(
        zx, m.conv_w, m.conv_b, xbc, m.B, m.Lseq, m.din, m.dinner, m.convdim);
    // 5) dt / da
    dtda_kernel<<<dim3(cdiv((long long)rows * m.H, WG)), dim3(WG), 0, stream>>>(
        zx, m.dt_bias, m.A_log, dt, da, rows, m.din, m.dinner + m.convdim, m.H);
    // 6) sequential scan
    scan_kernel<<<dim3(m.B * m.H), dim3(WG), 0, stream>>>(
        xbc, dt, da, m.Dh, ybuf, m.B, m.Lseq, m.H, m.convdim, m.dinner);
    // 7) gate + RMSNorm + split
    gate_rms_split_kernel<<<dim3(rows), dim3(WG), 0, stream>>>(
        zx, ybuf, m.rms_w, yn_hi, yn_lo, m.din, m.dinner);
    // 8) split W_out transposed (padded to NpadOut rows)
    split_wT_kernel<<<dim3(cdiv((long long)NpadOut * m.dinner, WG)), dim3(WG), 0, stream>>>(
        m.W_out, wo_hi, wo_lo, m.dinner, m.dmodel, NpadOut);
    // 9) out = y_norm @ W_out (+ residual)  (M=rows, N=dmodel, K=dinner)
    launch_gemm(m.dinner, yn_hi, yn_lo, wo_hi, wo_lo, dst, resid, rows, m.dmodel, NpadOut / STRIP);
  };

  // ---- Phase 1: spatial mamba on (B,L,C), residual = x -> x_spa_final ----
  off = phase_base;
  run_mamba(spa, x, x_spa_final, x);

  // ---- Phase 2: spectral mamba on transpose (B,C,L) ----
  off = phase_base;  // reuse phase-1 scratch
  float* xT   = (float*)alloc(NX * 4);
  float* gout = (float*)alloc(NX * 4);
  transpose_kernel<<<dim3(cdiv((long long)NX, WG)), dim3(WG), 0, stream>>>(
      x_spa_final, xT, BTOT, LTOT, CTOT);
  run_mamba(spe, xT, gout, nullptr);
  // out[b,l,c] = gout[b,c,l] + x_spa_final[b,l,c]
  transpose_add_kernel<<<dim3(cdiv((long long)NX, WG)), dim3(WG), 0, stream>>>(
      gout, x_spa_final, (float*)d_out, BTOT, LTOT, CTOT);
}
